// LongShortTermTransformerBlock_6914897346613
// MI455X (gfx1250) — compile-verified, matrix operand order checked by reading, not run on hardware
//
#include <hip/hip_runtime.h>
#include <hip/hip_bf16.h>
#include <math.h>

// ---- dims ----
constexpr int kD   = 256;
constexpr int kN   = 8;
constexpr int kH   = 30;
constexpr int kW   = 30;
constexpr int kHW  = 900;
constexpr int kDFF = 1024;
constexpr int kLMEM= 1800;
constexpr int kNH  = 8;     // heads (both self and attn branches)
constexpr int kHD  = 32;    // head dim
constexpr int kGN  = 32;    // groupnorm groups
constexpr int kP2  = 225;   // 15x15 window
constexpr int kTOK = kHW * kN;          // 7200 rows in seq-first layout
constexpr int kND  = kN * kD;           // token stride (elements)

typedef __attribute__((ext_vector_type(16))) _Float16 v16h;
typedef __attribute__((ext_vector_type(8)))  _Float16 v8h;
typedef __attribute__((ext_vector_type(8)))  float    v8f;

union AF { v16h v; v8h h[2]; };
union BF { v16h v; _Float16 e[16]; };

__device__ __forceinline__ int imin(int a, int b) { return a < b ? a : b; }

// ---------------------------------------------------------------------------
// f32 -> f16 elementwise convert
// ---------------------------------------------------------------------------
__global__ void cvt_h_kernel(const float* __restrict__ X, _Float16* __restrict__ Y, int n) {
  int i = blockIdx.x * 256 + threadIdx.x;
  if (i < n) Y[i] = (_Float16)X[i];
}

// ---------------------------------------------------------------------------
// transpose + convert: W[K,Nc] f32 -> Wt[Nc,K] f16
// ---------------------------------------------------------------------------
__global__ void transpose_h_kernel(const float* __restrict__ Wm, _Float16* __restrict__ Wt,
                                   int K, int Nc) {
  int idx = blockIdx.x * 256 + threadIdx.x;
  if (idx < K * Nc) {
    int i = idx / Nc, j = idx % Nc;
    Wt[(size_t)j * K + i] = (_Float16)Wm[idx];
  }
}

// ---------------------------------------------------------------------------
// LayerNorm over last dim (256); one block (256 thr) per row.
// outT = LN(x)*g+b (f16); if pos != null, outQK = outT + pos (f16)
// ---------------------------------------------------------------------------
__global__ __launch_bounds__(256) void ln_kernel(
    const float* __restrict__ x, const float* __restrict__ g, const float* __restrict__ b,
    const float* __restrict__ pos, _Float16* __restrict__ outT, _Float16* __restrict__ outQK) {
  int row = blockIdx.x, c = threadIdx.x;
  int wave = c >> 5, lane = c & 31;
  __shared__ float red[8];
  float v = x[(size_t)row * kD + c];
  float s = v;
  #pragma unroll
  for (int m = 1; m < 32; m <<= 1) s += __shfl_xor(s, m, 32);
  if (lane == 0) red[wave] = s;
  __syncthreads();
  float tot = 0.f;
  #pragma unroll
  for (int i = 0; i < 8; ++i) tot += red[i];
  float mean = tot * (1.0f / kD);
  float d = v - mean;
  float s2 = d * d;
  #pragma unroll
  for (int m = 1; m < 32; m <<= 1) s2 += __shfl_xor(s2, m, 32);
  __syncthreads();
  if (lane == 0) red[wave] = s2;
  __syncthreads();
  float tot2 = 0.f;
  #pragma unroll
  for (int i = 0; i < 8; ++i) tot2 += red[i];
  float rstd = rsqrtf(tot2 * (1.0f / kD) + 1e-5f);
  float y = d * rstd * g[c] + b[c];
  outT[(size_t)row * kD + c] = (_Float16)y;
  if (outQK) outQK[(size_t)row * kD + c] = (_Float16)(y + pos[(size_t)row * kD + c]);
}

// ---------------------------------------------------------------------------
// WMMA GEMM: out[M,Nc] = A[M,K](f16) @ Wt[Nc,K]^T(f16) + bias (+res)
// 4 waves per block; wave computes 16 rows x 64 cols. grid=(Nc/256, M/16).
// The 16x32 A tile is DMA'd into LDS once per block (double-buffered) with
// global_load_async_to_lds_b128 (ASYNCcnt), then each wave reads its WMMA
// A-fragment from LDS; B tiles stream from global as b128 loads.
// ---------------------------------------------------------------------------
__global__ __launch_bounds__(128) void gemm_wmma_kernel(
    const _Float16* __restrict__ A, const _Float16* __restrict__ Wt,
    const float* __restrict__ bias, const float* __restrict__ res,
    float* __restrict__ outF, _Float16* __restrict__ outH,
    int K, int Nc) {
  __shared__ __align__(16) _Float16 Asm[2][16 * 32];
  int tid = threadIdx.x;
  int lane = tid & 31, wave = tid >> 5;
  int half = lane >> 4, lm = lane & 15;
  int row0 = blockIdx.y * 16;
  int col0 = blockIdx.x * 256 + wave * 64;

  // async-DMA one 16x32 f16 A tile into LDS buffer `buf` (64 lanes x 16B)
  auto stage = [&](int kb, int buf) {
    if (tid < 64) {
      int row = tid >> 2;      // 0..15
      int seg = tid & 3;       // 4 x 8-half segments per row
      const _Float16* gp = A + (size_t)(row0 + row) * K + kb + seg * 8;
      unsigned lofs = (unsigned)(unsigned long long)&Asm[buf][row * 32 + seg * 8];
      asm volatile("global_load_async_to_lds_b128 %0, %1, off"
                   :: "v"(lofs), "v"((unsigned long long)gp)
                   : "memory");
    }
  };

  v8f acc0 = {}, acc1 = {}, acc2 = {}, acc3 = {};
  const _Float16* W0 = Wt + (size_t)(col0 + 0 * 16 + lm) * K;
  const _Float16* W1 = Wt + (size_t)(col0 + 1 * 16 + lm) * K;
  const _Float16* W2 = Wt + (size_t)(col0 + 2 * 16 + lm) * K;
  const _Float16* W3 = Wt + (size_t)(col0 + 3 * 16 + lm) * K;

  stage(0, 0);
  int buf = 0;
  for (int kb = 0; kb < K; kb += 32) {
    if (kb + 32 < K) {
      stage(kb + 32, buf ^ 1);
      asm volatile("s_wait_asynccnt 0x1" ::: "memory");
    } else {
      asm volatile("s_wait_asynccnt 0x0" ::: "memory");
    }
    __syncthreads();  // A tile `buf` visible to all 4 waves
    AF a;
    a.h[0] = *(const v8h*)(&Asm[buf][lm * 32 + half * 8]);
    a.h[1] = *(const v8h*)(&Asm[buf][lm * 32 + 16 + half * 8]);
    __syncthreads();  // all reads of `buf` done before next DMA overwrites it
    __builtin_prefetch(W0 + kb + 64, 0, 3);
    AF b0, b1, b2, b3;
    b0.h[0] = *(const v8h*)(W0 + kb + half * 8);
    b0.h[1] = *(const v8h*)(W0 + kb + 16 + half * 8);
    b1.h[0] = *(const v8h*)(W1 + kb + half * 8);
    b1.h[1] = *(const v8h*)(W1 + kb + 16 + half * 8);
    b2.h[0] = *(const v8h*)(W2 + kb + half * 8);
    b2.h[1] = *(const v8h*)(W2 + kb + 16 + half * 8);
    b3.h[0] = *(const v8h*)(W3 + kb + half * 8);
    b3.h[1] = *(const v8h*)(W3 + kb + 16 + half * 8);
    acc0 = __builtin_amdgcn_wmma_f32_16x16x32_f16(false, a.v, false, b0.v, (short)0, acc0, false, false);
    acc1 = __builtin_amdgcn_wmma_f32_16x16x32_f16(false, a.v, false, b1.v, (short)0, acc1, false, false);
    acc2 = __builtin_amdgcn_wmma_f32_16x16x32_f16(false, a.v, false, b2.v, (short)0, acc2, false, false);
    acc3 = __builtin_amdgcn_wmma_f32_16x16x32_f16(false, a.v, false, b3.v, (short)0, acc3, false, false);
    buf ^= 1;
  }
  v8f accs[4] = {acc0, acc1, acc2, acc3};
  #pragma unroll
  for (int t = 0; t < 4; ++t) {
    int col = col0 + t * 16 + lm;
    float bv = bias ? bias[col] : 0.f;
    #pragma unroll
    for (int r = 0; r < 8; ++r) {
      int row = row0 + r + 8 * half;
      size_t idx = (size_t)row * Nc + col;
      float v = accs[t][r] + bv;
      if (res) v += res[idx];
      if (outF) outF[idx] = v;
      if (outH) outH[idx] = (_Float16)v;
    }
  }
}

// ---------------------------------------------------------------------------
// Flash attention (1 wave per 16-query tile per (batch,head)).
// Q/K/V/O are f16 in [L, N, D] seq-first layout; head slice at n*D + h*32.
// ---------------------------------------------------------------------------
__global__ __launch_bounds__(32) void flash_attn_kernel(
    const _Float16* __restrict__ Q, const _Float16* __restrict__ Kk,
    const _Float16* __restrict__ Vv, _Float16* __restrict__ O,
    int Lq, int Lk, float scale) {
  __shared__ __align__(16) _Float16 P[16 * 32];
  int lane = threadIdx.x;
  int half = lane >> 4, lm = lane & 15;
  int qt = blockIdx.x;
  int n = blockIdx.y >> 3, h = blockIdx.y & 7;
  int base = n * kD + h * kHD;

  int qrow = imin(qt * 16 + lm, Lq - 1);
  const _Float16* qp = Q + (size_t)qrow * kND + base;
  AF qa;
  qa.h[0] = *(const v8h*)(qp + half * 8);
  qa.h[1] = *(const v8h*)(qp + 16 + half * 8);

  float rm[8], rs[8];
  v8f o0 = {}, o1 = {};
  #pragma unroll
  for (int r = 0; r < 8; ++r) { rm[r] = -3e30f; rs[r] = 0.f; }

  for (int t0 = 0; t0 < Lk; t0 += 32) {
    v8f s0, s1;
    {
      int tok = imin(t0 + lm, Lk - 1);
      const _Float16* kp = Kk + (size_t)tok * kND + base;
      AF kb; kb.h[0] = *(const v8h*)(kp + half * 8); kb.h[1] = *(const v8h*)(kp + 16 + half * 8);
      v8f z = {};
      s0 = __builtin_amdgcn_wmma_f32_16x16x32_f16(false, qa.v, false, kb.v, (short)0, z, false, false);
    }
    {
      int tok = imin(t0 + 16 + lm, Lk - 1);
      const _Float16* kp = Kk + (size_t)tok * kND + base;
      AF kb; kb.h[0] = *(const v8h*)(kp + half * 8); kb.h[1] = *(const v8h*)(kp + 16 + half * 8);
      v8f z = {};
      s1 = __builtin_amdgcn_wmma_f32_16x16x32_f16(false, qa.v, false, kb.v, (short)0, z, false, false);
    }
    bool ok0 = (t0 + lm) < Lk;
    bool ok1 = (t0 + 16 + lm) < Lk;
    #pragma unroll
    for (int r = 0; r < 8; ++r) {
      float a0 = ok0 ? s0[r] * scale : -3e30f;
      float a1 = ok1 ? s1[r] * scale : -3e30f;
      float mx = fmaxf(a0, a1);
      #pragma unroll
      for (int m = 1; m < 16; m <<= 1) mx = fmaxf(mx, __shfl_xor(mx, m, 32));
      float nm = fmaxf(rm[r], mx);
      float f = __expf(rm[r] - nm);
      float p0 = __expf(a0 - nm), p1 = __expf(a1 - nm);
      float ts = p0 + p1;
      #pragma unroll
      for (int m = 1; m < 16; m <<= 1) ts += __shfl_xor(ts, m, 32);
      rs[r] = rs[r] * f + ts;
      rm[r] = nm;
      o0[r] *= f; o1[r] *= f;
      int prow = r + 8 * half;
      P[prow * 32 + lm]      = (_Float16)p0;
      P[prow * 32 + 16 + lm] = (_Float16)p1;
    }
    __syncthreads();
    AF pa;
    pa.h[0] = *(const v8h*)(&P[lm * 32 + half * 8]);
    pa.h[1] = *(const v8h*)(&P[lm * 32 + 16 + half * 8]);
    BF vb0, vb1;
    #pragma unroll
    for (int e = 0; e < 16; ++e) {
      int kk = (e >> 3) * 16 + half * 8 + (e & 7);
      int tok = imin(t0 + kk, Lk - 1);
      const _Float16* vp = Vv + (size_t)tok * kND + base;
      vb0.e[e] = vp[lm];
      vb1.e[e] = vp[16 + lm];
    }
    o0 = __builtin_amdgcn_wmma_f32_16x16x32_f16(false, pa.v, false, vb0.v, (short)0, o0, false, false);
    o1 = __builtin_amdgcn_wmma_f32_16x16x32_f16(false, pa.v, false, vb1.v, (short)0, o1, false, false);
    __syncthreads();
  }
  #pragma unroll
  for (int r = 0; r < 8; ++r) {
    int row = qt * 16 + r + 8 * half;
    if (row < Lq) {
      float inv = 1.0f / rs[r];
      _Float16* op = O + (size_t)row * kND + base;
      op[lm]      = (_Float16)(o0[r] * inv);
      op[16 + lm] = (_Float16)(o1[r] * inv);
    }
  }
}

// ---------------------------------------------------------------------------
// Local 15x15 window attention: one wave per (n, head, y, x); lane = channel.
// score_p = sum_d q_d*(k_d/sqrt(HD) + relW[g,p,d]) + relB[g,p]; zero-pad OOB K/V.
// ---------------------------------------------------------------------------
__global__ __launch_bounds__(256) void local_attn_kernel(
    const float* __restrict__ Qf, const float* __restrict__ Kf, const float* __restrict__ Vf,
    const float* __restrict__ relw, const float* __restrict__ relb, _Float16* __restrict__ out) {
  int wid = blockIdx.x * 8 + (threadIdx.x >> 5);
  int lane = threadIdx.x & 31;
  int pos = wid % kHW;
  int g = (wid / kHW) & 7;
  int n = wid / (kHW * kNH);
  int y = pos / kW, x = pos % kW;
  const float invs = 0.1767766953f; // 1/sqrt(32)

  float q = Qf[((size_t)pos * kN + n) * kD + g * kHD + lane];
  const float* kc = Kf + ((size_t)n * kD + g * kHD + lane) * kHW;
  const float* vc = Vf + ((size_t)n * kD + g * kHD + lane) * kHW;
  const float* rw = relw + (size_t)g * kP2 * kHD;

  float sc[8];
  float mx = -3e30f;
  #pragma unroll
  for (int slot = 0; slot < 8; ++slot) {
    float keep = 0.f;
    for (int j = 0; j < 32; ++j) {
      int p = slot * 32 + j;
      float s = 0.f;
      if (p < kP2) {
        int dy = p / 15 - 7, dx = p % 15 - 7;
        int yy = y + dy, xx = x + dx;
        float kv = 0.f;
        if (yy >= 0 && yy < kH && xx >= 0 && xx < kW) kv = kc[yy * kW + xx];
        float part = q * (kv * invs + rw[p * kHD + lane]);
        #pragma unroll
        for (int m = 1; m < 32; m <<= 1) part += __shfl_xor(part, m, 32);
        s = part + relb[g * kP2 + p];
        mx = fmaxf(mx, s);
      }
      if (j == lane) keep = s;
    }
    sc[slot] = keep;
  }
  float e[8];
  float sum = 0.f;
  #pragma unroll
  for (int slot = 0; slot < 8; ++slot) {
    int p = slot * 32 + lane;
    e[slot] = (p < kP2) ? __expf(sc[slot] - mx) : 0.f;
    sum += e[slot];
  }
  #pragma unroll
  for (int m = 1; m < 32; m <<= 1) sum += __shfl_xor(sum, m, 32);
  float inv = 1.f / sum;
  float o = 0.f;
  #pragma unroll
  for (int slot = 0; slot < 8; ++slot) {
    for (int j = 0; j < 32; ++j) {
      int p = slot * 32 + j;
      if (p >= kP2) break;
      float a = __shfl(e[slot], j, 32) * inv;
      int dy = p / 15 - 7, dx = p % 15 - 7;
      int yy = y + dy, xx = x + dx;
      if (yy >= 0 && yy < kH && xx >= 0 && xx < kW) o += a * vc[yy * kW + xx];
    }
  }
  out[((size_t)pos * kN + n) * kD + g * kHD + lane] = (_Float16)o;
}

// ---------------------------------------------------------------------------
// GroupNorm stats over (32ch x 900 pix) per (batch, group). h1 is [hw*n, DFF].
// ---------------------------------------------------------------------------
__global__ __launch_bounds__(256) void gn_stats_kernel(const float* __restrict__ h1,
                                                       float* __restrict__ stats) {
  int n = blockIdx.x >> 5, g = blockIdx.x & 31;
  float s = 0.f, s2 = 0.f;
  for (int i = threadIdx.x; i < 32 * kHW; i += 256) {
    int cc = i & 31, yx = i >> 5;
    float v = h1[((size_t)yx * kN + n) * kDFF + g * 32 + cc];
    s += v; s2 += v * v;
  }
  #pragma unroll
  for (int m = 1; m < 32; m <<= 1) { s += __shfl_xor(s, m, 32); s2 += __shfl_xor(s2, m, 32); }
  __shared__ float r1[8], r2[8];
  int wave = threadIdx.x >> 5, lane = threadIdx.x & 31;
  if (lane == 0) { r1[wave] = s; r2[wave] = s2; }
  __syncthreads();
  if (threadIdx.x == 0) {
    float t = 0.f, t2 = 0.f;
    #pragma unroll
    for (int i = 0; i < 8; ++i) { t += r1[i]; t2 += r2[i]; }
    float cnt = 32.f * kHW;
    float mean = t / cnt;
    float var = t2 / cnt - mean * mean;
    stats[blockIdx.x * 2] = mean;
    stats[blockIdx.x * 2 + 1] = rsqrtf(var + 1e-5f);
  }
}

// ---------------------------------------------------------------------------
// GN apply + exact GELU; rearrange [hw,n,dff] -> [n,dff,hw] for the dwconv.
// ---------------------------------------------------------------------------
__global__ void gn_act_kernel(const float* __restrict__ h1, const float* __restrict__ stats,
                              const float* __restrict__ gg, const float* __restrict__ gb,
                              float* __restrict__ act) {
  int idx = blockIdx.x * 256 + threadIdx.x;
  if (idx >= kN * kDFF * kHW) return;
  int yx = idx % kHW;
  int c = (idx / kHW) % kDFF;
  int n = idx / (kHW * kDFF);
  float v = h1[((size_t)yx * kN + n) * kDFF + c];
  int g = c >> 5;
  float mean = stats[(n * 32 + g) * 2];
  float rstd = stats[(n * 32 + g) * 2 + 1];
  v = (v - mean) * rstd * gg[c] + gb[c];
  v = 0.5f * v * (1.f + erff(v * 0.70710678118f));
  act[idx] = v;
}

// ---------------------------------------------------------------------------
// 5x5 depthwise conv (pad 2, no bias); writes f16 [hw,n,dff] for final GEMM.
// ---------------------------------------------------------------------------
__global__ void dwconv_kernel(const float* __restrict__ act, const float* __restrict__ dw,
                              _Float16* __restrict__ out) {
  int idx = blockIdx.x * 256 + threadIdx.x;
  if (idx >= kN * kDFF * kHW) return;
  int yx = idx % kHW;
  int c = (idx / kHW) % kDFF;
  int n = idx / (kHW * kDFF);
  int y = yx / kW, x = yx % kW;
  const float* ap = act + ((size_t)n * kDFF + c) * kHW;
  const float* wp = dw + c * 25;
  float acc = 0.f;
  #pragma unroll
  for (int dy = 0; dy < 5; ++dy) {
    int yy = y + dy - 2;
    if (yy < 0 || yy >= kH) continue;
    #pragma unroll
    for (int dx = 0; dx < 5; ++dx) {
      int xx = x + dx - 2;
      if (xx < 0 || xx >= kW) continue;
      acc += ap[yy * kW + xx] * wp[dy * 5 + dx];
    }
  }
  out[((size_t)yx * kN + n) * kDFF + c] = (_Float16)acc;
}

// ---------------------------------------------------------------------------
// host orchestration
// ---------------------------------------------------------------------------
extern "C" void kernel_launch(void* const* d_in, const int* in_sizes, int n_in,
                              void* d_out, int out_size, void* d_ws, size_t ws_size,
                              hipStream_t stream) {
  (void)in_sizes; (void)n_in; (void)out_size; (void)ws_size;
  // inputs
  const float* tgt      = (const float*)d_in[0];
  const float* global_K = (const float*)d_in[1];
  const float* global_V = (const float*)d_in[2];
  const float* local_K  = (const float*)d_in[3];
  const float* local_V  = (const float*)d_in[4];
  const float* self_pos = (const float*)d_in[5];
  // params (dict order)
  const float* ln1_g = (const float*)d_in[6];   const float* ln1_b = (const float*)d_in[7];
  const float* ln2_g = (const float*)d_in[8];   const float* ln2_b = (const float*)d_in[9];
  const float* ln3_g = (const float*)d_in[10];  const float* ln3_b = (const float*)d_in[11];
  const float* sa_qw = (const float*)d_in[12];  const float* sa_qb = (const float*)d_in[13];
  const float* sa_kw = (const float*)d_in[14];  const float* sa_kb = (const float*)d_in[15];
  const float* sa_vw = (const float*)d_in[16];  const float* sa_vb = (const float*)d_in[17];
  const float* sa_ow = (const float*)d_in[18];  const float* sa_ob = (const float*)d_in[19];
  const float* q_w   = (const float*)d_in[20];  const float* q_b   = (const float*)d_in[21];
  const float* lt_ow = (const float*)d_in[22];  const float* lt_ob = (const float*)d_in[23];
  const float* st_relw = (const float*)d_in[24];
  const float* st_relb = (const float*)d_in[25];
  const float* st_ow = (const float*)d_in[26];  const float* st_ob = (const float*)d_in[27];
  const float* w1    = (const float*)d_in[28];  const float* b1    = (const float*)d_in[29];
  const float* gn_g  = (const float*)d_in[30];  const float* gn_b  = (const float*)d_in[31];
  const float* dw_w  = (const float*)d_in[32];
  const float* w2    = (const float*)d_in[33];  const float* b2    = (const float*)d_in[34];

  // bump allocator over d_ws
  char* ws = (char*)d_ws;
  size_t off = 0;
  auto alloc = [&](size_t bytes) -> void* {
    off = (off + 255) & ~(size_t)255;
    void* p = ws + off;
    off += bytes;
    return p;
  };
  const size_t SZ_DD  = (size_t)kD * kD;
  const size_t SZ_TOK = (size_t)kTOK * kD;     // 7200*256
  const size_t SZ_MEM = (size_t)kLMEM * kND;   // 1800*2048
  const size_t SZ_FF  = (size_t)kTOK * kDFF;   // 7200*1024

  _Float16* WqT  = (_Float16*)alloc(SZ_DD * 2);
  _Float16* WkT  = (_Float16*)alloc(SZ_DD * 2);
  _Float16* WvT  = (_Float16*)alloc(SZ_DD * 2);
  _Float16* WoT  = (_Float16*)alloc(SZ_DD * 2);
  _Float16* WquT = (_Float16*)alloc(SZ_DD * 2);
  _Float16* WltT = (_Float16*)alloc(SZ_DD * 2);
  _Float16* WstT = (_Float16*)alloc(SZ_DD * 2);
  _Float16* W1T  = (_Float16*)alloc((size_t)kD * kDFF * 2);
  _Float16* W2T  = (_Float16*)alloc((size_t)kDFF * kD * 2);
  _Float16* gK16 = (_Float16*)alloc(SZ_MEM * 2);
  _Float16* gV16 = (_Float16*)alloc(SZ_MEM * 2);
  _Float16* t16    = (_Float16*)alloc(SZ_TOK * 2);
  _Float16* qkin16 = (_Float16*)alloc(SZ_TOK * 2);
  _Float16* sq16   = (_Float16*)alloc(SZ_TOK * 2);
  _Float16* sk16   = (_Float16*)alloc(SZ_TOK * 2);
  _Float16* sv16   = (_Float16*)alloc(SZ_TOK * 2);
  _Float16* aO16   = (_Float16*)alloc(SZ_TOK * 2);
  float*    tgt1   = (float*)alloc(SZ_TOK * 4);
  float*    currQf = (float*)alloc(SZ_TOK * 4);
  _Float16* currQ16= (_Float16*)alloc(SZ_TOK * 2);
  _Float16* ltO16  = (_Float16*)alloc(SZ_TOK * 2);
  _Float16* locO16 = (_Float16*)alloc(SZ_TOK * 2);
  float*    tmidA  = (float*)alloc(SZ_TOK * 4);
  float*    tmidB  = (float*)alloc(SZ_TOK * 4);
  float*    h1     = (float*)alloc(SZ_FF * 4);
  float*    stats  = (float*)alloc(512 * 4);
  float*    act    = (float*)alloc(SZ_FF * 4);
  _Float16* h2_16  = (_Float16*)alloc(SZ_FF * 2);

  const float scale = 0.1767766953f; // 1/sqrt(32)

  // 1) weight prep
  transpose_h_kernel<<<(int)((SZ_DD + 255) / 256), 256, 0, stream>>>(sa_qw, WqT, kD, kD);
  transpose_h_kernel<<<(int)((SZ_DD + 255) / 256), 256, 0, stream>>>(sa_kw, WkT, kD, kD);
  transpose_h_kernel<<<(int)((SZ_DD + 255) / 256), 256, 0, stream>>>(sa_vw, WvT, kD, kD);
  transpose_h_kernel<<<(int)((SZ_DD + 255) / 256), 256, 0, stream>>>(sa_ow, WoT, kD, kD);
  transpose_h_kernel<<<(int)((SZ_DD + 255) / 256), 256, 0, stream>>>(q_w,   WquT, kD, kD);
  transpose_h_kernel<<<(int)((SZ_DD + 255) / 256), 256, 0, stream>>>(lt_ow, WltT, kD, kD);
  transpose_h_kernel<<<(int)((SZ_DD + 255) / 256), 256, 0, stream>>>(st_ow, WstT, kD, kD);
  transpose_h_kernel<<<(kD * kDFF + 255) / 256, 256, 0, stream>>>(w1, W1T, kD, kDFF);
  transpose_h_kernel<<<(kDFF * kD + 255) / 256, 256, 0, stream>>>(w2, W2T, kDFF, kD);
  cvt_h_kernel<<<(int)((SZ_MEM + 255) / 256), 256, 0, stream>>>(global_K, gK16, (int)SZ_MEM);
  cvt_h_kernel<<<(int)((SZ_MEM + 255) / 256), 256, 0, stream>>>(global_V, gV16, (int)SZ_MEM);

  // 2) self-attention branch
  ln_kernel<<<kTOK, 256, 0, stream>>>(tgt, ln1_g, ln1_b, self_pos, t16, qkin16);
  gemm_wmma_kernel<<<dim3(1, kTOK / 16), 128, 0, stream>>>(qkin16, WqT, sa_qb, nullptr, nullptr, sq16, kD, kD);
  gemm_wmma_kernel<<<dim3(1, kTOK / 16), 128, 0, stream>>>(qkin16, WkT, sa_kb, nullptr, nullptr, sk16, kD, kD);
  gemm_wmma_kernel<<<dim3(1, kTOK / 16), 128, 0, stream>>>(t16,    WvT, sa_vb, nullptr, nullptr, sv16, kD, kD);
  flash_attn_kernel<<<dim3((kHW + 15) / 16, kN * kNH), 32, 0, stream>>>(sq16, sk16, sv16, aO16, kHW, kHW, scale);
  gemm_wmma_kernel<<<dim3(1, kTOK / 16), 128, 0, stream>>>(aO16, WoT, sa_ob, tgt, tgt1, nullptr, kD, kD);

  // 3) long-term memory attention
  ln_kernel<<<kTOK, 256, 0, stream>>>(tgt1, ln2_g, ln2_b, nullptr, t16, nullptr);
  gemm_wmma_kernel<<<dim3(1, kTOK / 16), 128, 0, stream>>>(t16, WquT, q_b, nullptr, currQf, currQ16, kD, kD);
  flash_attn_kernel<<<dim3((kHW + 15) / 16, kN * kNH), 32, 0, stream>>>(currQ16, gK16, gV16, ltO16, kHW, kLMEM, scale);
  gemm_wmma_kernel<<<dim3(1, kTOK / 16), 128, 0, stream>>>(ltO16, WltT, lt_ob, tgt1, tmidA, nullptr, kD, kD);

  // 4) short-term (local window) attention
  local_attn_kernel<<<kN * kNH * kHW / 8, 256, 0, stream>>>(currQf, local_K, local_V, st_relw, st_relb, locO16);
  gemm_wmma_kernel<<<dim3(1, kTOK / 16), 128, 0, stream>>>(locO16, WstT, st_ob, tmidA, tmidB, nullptr, kD, kD);

  // 5) gated depthwise-conv FFN
  ln_kernel<<<kTOK, 256, 0, stream>>>(tmidB, ln3_g, ln3_b, nullptr, t16, nullptr);
  gemm_wmma_kernel<<<dim3(kDFF / 256, kTOK / 16), 128, 0, stream>>>(t16, W1T, b1, nullptr, h1, nullptr, kD, kDFF);
  gn_stats_kernel<<<kN * kGN, 256, 0, stream>>>(h1, stats);
  gn_act_kernel<<<(int)((SZ_FF + 255) / 256), 256, 0, stream>>>(h1, stats, gn_g, gn_b, act);
  dwconv_kernel<<<(int)((SZ_FF + 255) / 256), 256, 0, stream>>>(act, dw_w, h2_16);
  gemm_wmma_kernel<<<dim3(1, kTOK / 16), 128, 0, stream>>>(h2_16, W2T, b2, tmidB, (float*)d_out, nullptr, kDFF, kD);
}
